// EdgeConv_82849919139973
// MI455X (gfx1250) — matrix operand order repaired and checked by reading, not compiled, not run on hardware
//
#include <hip/hip_runtime.h>
#include <hip/hip_bf16.h>
#include <stdint.h>

// ---------------------------------------------------------------------------
// Problem constants (from reference): B=4, N=2048, E=16384, ND=256, ED=128,
// TD=64, C=512, H=8, DH=64.  L = E+N edges (self loops appended).
// ---------------------------------------------------------------------------
#define B_    4
#define N_    2048
#define E_    16384
#define ND_   256
#define ED_   128
#define TD_   64
#define C_    512
#define H_    8
#define DH_   64
#define LED_  (E_ + N_)          // 18432 edges incl. self loops
#define MT_   (B_ * LED_)        // 73728 token rows
#define EPS_  1e-5f

typedef __attribute__((ext_vector_type(16))) _Float16 v16h;
typedef __attribute__((ext_vector_type(8)))  _Float16 v8h;
typedef __attribute__((ext_vector_type(8)))  float    v8f;

union V16 { v16h v; v8h h[2]; };

enum { EPI_F32 = 1, EPI_F16 = 2, EPI_RES = 4, EPI_GELU = 8 };

// ---------------------------------------------------------------------------
// fp32 -> fp16 weight conversion
// ---------------------------------------------------------------------------
__global__ __launch_bounds__(256)
void k_f32_to_f16(const float* __restrict__ s, _Float16* __restrict__ d, int n) {
    int i = blockIdx.x * 256 + threadIdx.x;
    if (i < n) d[i] = (_Float16)s[i];
}

// ---------------------------------------------------------------------------
// Gather + input feature construction.
// One block (256 threads) per edge-row.
// ---------------------------------------------------------------------------
__global__ __launch_bounds__(256)
void k_build_inputs(const float* __restrict__ x, const float* __restrict__ ea,
                    const float* __restrict__ ts,
                    const float* __restrict__ Wtemb, const float* __restrict__ btemb,
                    const int* __restrict__ eidx,
                    _Float16* __restrict__ nodecat, _Float16* __restrict__ timecat,
                    _Float16* __restrict__ edgein) {
    int row = blockIdx.x;                 // 0..MT_-1
    int b   = row / LED_;
    int l   = row - b * LED_;
    int src, tgt;
    if (l < E_) {
        src = eidx[((size_t)b * 2 + 0) * E_ + l];
        tgt = eidx[((size_t)b * 2 + 1) * E_ + l];
    } else {
        src = tgt = l - E_;               // self loop
    }
    int t = threadIdx.x;                  // 0..255
    // node features: x_i || x_j
    nodecat[(size_t)row * 512 + t]       = (_Float16)x[((size_t)b * N_ + src) * ND_ + t];
    nodecat[(size_t)row * 512 + 256 + t] = (_Float16)x[((size_t)b * N_ + tgt) * ND_ + t];
    if (t < 128) {
        // time encoding: sin(t * W_temb + b_temb), i then j
        int which = t >> 6;               // 0 -> src, 1 -> tgt
        int c     = t & 63;
        float tv  = ts[(size_t)b * N_ + (which ? tgt : src)];
        timecat[(size_t)row * 128 + t] = (_Float16)sinf(tv * Wtemb[c] + btemb[c]);
        // edge features (zero for appended self loops)
        float ev = (l < E_) ? ea[((size_t)b * E_ + l) * ED_ + t] : 0.0f;
        edgein[(size_t)row * 128 + t] = (_Float16)ev;
    }
}

// ---------------------------------------------------------------------------
// WMMA GEMM:  out(M,N) = A(M,K) * W(N,K)^T + bias   (fp16 in, f32 accumulate)
// Block = 128 threads = 4 waves; each wave owns a 16(M) x 64(N) tile.
// All 4 waves share the same 16-row A tile, so A is staged Global->LDS with
// CDNA5 async copies (GLOBAL_LOAD_ASYNC_TO_LDS_B128, ASYNCcnt) and double
// buffered in 64-wide K chunks; fragments are read back with ds_load_b128.
// B fragments come straight from global (weights are L2-resident).
// grid = (N/256, M/16).
// A 16x32 / B 32x16 fragments use contiguous k = 16*(lane/16) + s; f32 C/D per CDNA5 ISA 7.12.2.
// ---------------------------------------------------------------------------
#define KCHUNK_  64
#define ASTRIDE_ 72   // padded LDS row stride in halves (144B, 16B-aligned)

template <int EPI>
__global__ __launch_bounds__(128)
void k_gemm_wmma(const _Float16* __restrict__ A, const _Float16* __restrict__ W,
                 const float* __restrict__ bias, int Ncols, int K,
                 float* __restrict__ outF, _Float16* __restrict__ outH,
                 const float* __restrict__ res) {
    __shared__ _Float16 sA[2][16 * ASTRIDE_];     // 2 x 2.25KB double buffer

    const int tid  = threadIdx.x;
    const int lane = tid & 31;
    const int wave = tid >> 5;                    // 0..3
    const int m0   = blockIdx.y * 16;
    const int n0   = blockIdx.x * 256 + wave * 64;
    const int lh   = lane & 15;
    const int kh   = lane >> 4;                   // 0/1

    // Async-copy mapping: thread t moves one 16B chunk of the 16x64 tile.
    // row = t/8, 8 chunks of 8 halves per row.
    const int arow_t = tid >> 3;                  // 0..15
    const int acol_t = (tid & 7) * 8;             // half offset within row
    const _Float16* agsrc = A + (size_t)(m0 + arow_t) * K + acol_t;

    auto issue_copy = [&](int buf, int k0) {
        uint32_t ldst =
            (uint32_t)(uintptr_t)(&sA[buf][arow_t * ASTRIDE_ + acol_t]);
        const _Float16* src = agsrc + k0;
        asm volatile("global_load_async_to_lds_b128 %0, %1, off"
                     :: "v"(ldst), "v"(src) : "memory");
    };

    issue_copy(0, 0);                             // prologue: stage chunk 0

    v8f acc[4];
#pragma unroll
    for (int t = 0; t < 4; ++t) { v8f z = {}; acc[t] = z; }

    int buf = 0;
    for (int k0 = 0; k0 < K; k0 += KCHUNK_) {
        bool more = (k0 + KCHUNK_) < K;
        if (more) issue_copy(buf ^ 1, k0 + KCHUNK_);   // overlap next chunk
        if (more) asm volatile("s_wait_asynccnt 1" ::: "memory");
        else      asm volatile("s_wait_asynccnt 0" ::: "memory");
        __syncthreads();                               // all waves' chunks in LDS
#pragma unroll
        for (int kk = 0; kk < KCHUNK_; kk += 32) {
            V16 a;
            a.h[0] = *(const v8h*)(&sA[buf][lh * ASTRIDE_ + kk + kh * 16]);
            a.h[1] = *(const v8h*)(&sA[buf][lh * ASTRIDE_ + kk + kh * 16 + 8]);
#pragma unroll
            for (int t = 0; t < 4; ++t) {
                const _Float16* wrow =
                    W + (size_t)(n0 + t * 16 + lh) * K + k0 + kk + kh * 16;
                __builtin_prefetch(wrow + KCHUNK_, 0, 1);   // global_prefetch_b8
                v16h bfrag = *(const v16h*)wrow;
                acc[t] = __builtin_amdgcn_wmma_f32_16x16x32_f16(
                    false, a.v, false, bfrag, (short)0, acc[t], false, false);
            }
        }
        __syncthreads();   // protect buf before next iteration's async overwrite
        buf ^= 1;
    }
    // epilogue: C/D layout -> lane%16 = N col, VGPR r -> M = r + 8*(lane/16)
#pragma unroll
    for (int t = 0; t < 4; ++t) {
        int   ncol = n0 + t * 16 + lh;
        float bv   = bias[ncol];
#pragma unroll
        for (int r = 0; r < 8; ++r) {
            int    mrow = m0 + r + 8 * kh;
            size_t idx  = (size_t)mrow * Ncols + ncol;
            float  val  = acc[t][r] + bv;
            if (EPI & EPI_RES)  val += res[idx];
            if (EPI & EPI_GELU) val  = 0.5f * val * (1.0f + erff(val * 0.70710678f));
            if (EPI & EPI_F32)  outF[idx] = val;
            if (EPI & EPI_F16)  outH[idx] = (_Float16)val;
        }
    }
}

// ---------------------------------------------------------------------------
// Batch-mixing attention: per (edge n, head h) a 4x4 score matrix across the
// batch dim l,m; softmax over m; ctx[l] = sum_m a[l][m] v[m].
// One wave per (n, h); lane owns 2 of DH=64 channels.
// ---------------------------------------------------------------------------
__global__ __launch_bounds__(256)
void k_edge_attention(const _Float16* __restrict__ q, const _Float16* __restrict__ k,
                      const _Float16* __restrict__ v, _Float16* __restrict__ ctx) {
    int n    = blockIdx.x;                // edge 0..LED_-1
    int h    = threadIdx.x >> 5;          // head 0..7
    int lane = threadIdx.x & 31;
    int d0   = lane * 2;

    float qf[B_][2], kf[B_][2], vf[B_][2];
#pragma unroll
    for (int l = 0; l < B_; ++l) {
        size_t base = ((size_t)l * LED_ + n) * C_ + h * DH_ + d0;
        qf[l][0] = (float)q[base]; qf[l][1] = (float)q[base + 1];
        kf[l][0] = (float)k[base]; kf[l][1] = (float)k[base + 1];
        vf[l][0] = (float)v[base]; vf[l][1] = (float)v[base + 1];
    }
    float s[B_][B_];
#pragma unroll
    for (int l = 0; l < B_; ++l)
#pragma unroll
        for (int m = 0; m < B_; ++m)
            s[l][m] = qf[l][0] * kf[m][0] + qf[l][1] * kf[m][1];
    // wave reduction over DH
#pragma unroll
    for (int off = 16; off > 0; off >>= 1)
#pragma unroll
        for (int l = 0; l < B_; ++l)
#pragma unroll
            for (int m = 0; m < B_; ++m)
                s[l][m] += __shfl_xor(s[l][m], off, 32);

    const float scale = 0.125f;           // 1/sqrt(DH)
    float a[B_][B_];
#pragma unroll
    for (int l = 0; l < B_; ++l) {
        float mx = -1e30f;
#pragma unroll
        for (int m = 0; m < B_; ++m) mx = fmaxf(mx, s[l][m] * scale);
        float sum = 0.0f;
#pragma unroll
        for (int m = 0; m < B_; ++m) { a[l][m] = expf(s[l][m] * scale - mx); sum += a[l][m]; }
        float inv = 1.0f / sum;
#pragma unroll
        for (int m = 0; m < B_; ++m) a[l][m] *= inv;
    }
#pragma unroll
    for (int l = 0; l < B_; ++l) {
        float c0 = 0.0f, c1 = 0.0f;
#pragma unroll
        for (int m = 0; m < B_; ++m) { c0 += a[l][m] * vf[m][0]; c1 += a[l][m] * vf[m][1]; }
        size_t base = ((size_t)l * LED_ + n) * C_ + h * DH_ + d0;
        ctx[base]     = (_Float16)c0;
        ctx[base + 1] = (_Float16)c1;
    }
}

// ---------------------------------------------------------------------------
// Row LayerNorm over C=512: one wave per row (16 elems/lane), in-place safe.
// ---------------------------------------------------------------------------
__global__ __launch_bounds__(256)
void k_layernorm(const float* __restrict__ in, const float* __restrict__ g,
                 const float* __restrict__ be, float* __restrict__ outF,
                 _Float16* __restrict__ outH) {
    int row  = blockIdx.x * 8 + (threadIdx.x >> 5);
    int lane = threadIdx.x & 31;
    const float* p = in + (size_t)row * C_;
    float vals[16];
    float mean = 0.0f;
#pragma unroll
    for (int i = 0; i < 16; ++i) { vals[i] = p[i * 32 + lane]; mean += vals[i]; }
#pragma unroll
    for (int off = 16; off > 0; off >>= 1) mean += __shfl_xor(mean, off, 32);
    mean *= (1.0f / C_);
    float var = 0.0f;
#pragma unroll
    for (int i = 0; i < 16; ++i) { float d = vals[i] - mean; var += d * d; }
#pragma unroll
    for (int off = 16; off > 0; off >>= 1) var += __shfl_xor(var, off, 32);
    var *= (1.0f / C_);
    float rs = rsqrtf(var + EPS_);
#pragma unroll
    for (int i = 0; i < 16; ++i) {
        int   c = i * 32 + lane;
        float o = (vals[i] - mean) * rs * g[c] + be[c];
        outF[(size_t)row * C_ + c] = o;
        if (outH) outH[(size_t)row * C_ + c] = (_Float16)o;
    }
}

// ---------------------------------------------------------------------------
// Host orchestration
// ---------------------------------------------------------------------------
extern "C" void kernel_launch(void* const* d_in, const int* in_sizes, int n_in,
                              void* d_out, int out_size, void* d_ws, size_t ws_size,
                              hipStream_t stream) {
    (void)in_sizes; (void)n_in; (void)out_size; (void)ws_size;

    const float* x        = (const float*)d_in[0];
    const float* edgeattr = (const float*)d_in[1];
    const float* ts       = (const float*)d_in[2];
    const float* W_node   = (const float*)d_in[3];
    const float* b_node   = (const float*)d_in[4];
    const float* W_temb   = (const float*)d_in[5];
    const float* b_temb   = (const float*)d_in[6];
    const float* W_timef  = (const float*)d_in[7];
    const float* b_timef  = (const float*)d_in[8];
    const float* W_edge   = (const float*)d_in[9];
    const float* b_edge   = (const float*)d_in[10];
    const float* Wq       = (const float*)d_in[11];
    const float* bq       = (const float*)d_in[12];
    const float* Wk       = (const float*)d_in[13];
    const float* bk       = (const float*)d_in[14];
    const float* Wv       = (const float*)d_in[15];
    const float* bv       = (const float*)d_in[16];
    const float* Wo       = (const float*)d_in[17];
    const float* bo       = (const float*)d_in[18];
    const float* W1       = (const float*)d_in[19];
    const float* b1       = (const float*)d_in[20];
    const float* W2       = (const float*)d_in[21];
    const float* b2       = (const float*)d_in[22];
    const float* g1       = (const float*)d_in[23];
    const float* be1      = (const float*)d_in[24];
    const float* g2       = (const float*)d_in[25];
    const float* be2      = (const float*)d_in[26];
    const int*   eidx     = (const int*)d_in[27];
    float*       out      = (float*)d_out;

    // --- workspace carve-out (256B aligned) ---
    char*  wsb = (char*)d_ws;
    size_t off = 0;
    auto take = [&](size_t bytes) -> void* {
        void* p = wsb + off;
        off += (bytes + 255) & ~(size_t)255;
        return p;
    };
    const size_t MT = MT_;
    _Float16* Wnode_h  = (_Float16*)take((size_t)512 * 512 * 2);
    _Float16* Wtimef_h = (_Float16*)take((size_t)512 * 128 * 2);
    _Float16* Wedge_h  = (_Float16*)take((size_t)512 * 128 * 2);
    _Float16* Wq_h     = (_Float16*)take((size_t)512 * 512 * 2);
    _Float16* Wk_h     = (_Float16*)take((size_t)512 * 512 * 2);
    _Float16* Wv_h     = (_Float16*)take((size_t)512 * 512 * 2);
    _Float16* Wo_h     = (_Float16*)take((size_t)512 * 512 * 2);
    _Float16* W1_h     = (_Float16*)take((size_t)2048 * 512 * 2);
    _Float16* W2_h     = (_Float16*)take((size_t)512 * 2048 * 2);
    _Float16* nodecat  = (_Float16*)take(MT * 512 * 2);
    _Float16* timecat  = (_Float16*)take(MT * 128 * 2);
    _Float16* edgein   = (_Float16*)take(MT * 128 * 2);
    float*    nodef_f  = (float*)   take(MT * 512 * 4);
    _Float16* nodef_h  = (_Float16*)take(MT * 512 * 2);
    _Float16* timef_h  = (_Float16*)take(MT * 512 * 2);
    _Float16* edgef_h  = (_Float16*)take(MT * 512 * 2);
    _Float16* q_h      = (_Float16*)take(MT * 512 * 2);
    _Float16* k_h      = (_Float16*)take(MT * 512 * 2);
    _Float16* v_h      = (_Float16*)take(MT * 512 * 2);
    _Float16* ctx_h    = (_Float16*)take(MT * 512 * 2);
    float*    out1_f   = (float*)   take(MT * 512 * 4);
    _Float16* out1_h   = (_Float16*)take(MT * 512 * 2);
    _Float16* ffn1_h   = (_Float16*)take(MT * 2048 * 2);

    // --- 1) convert weights to fp16 ---
    auto cvt = [&](const float* s, _Float16* d, int n) {
        k_f32_to_f16<<<(n + 255) / 256, 256, 0, stream>>>(s, d, n);
    };
    cvt(W_node,  Wnode_h,  512 * 512);
    cvt(W_timef, Wtimef_h, 512 * 128);
    cvt(W_edge,  Wedge_h,  512 * 128);
    cvt(Wq, Wq_h, 512 * 512);
    cvt(Wk, Wk_h, 512 * 512);
    cvt(Wv, Wv_h, 512 * 512);
    cvt(Wo, Wo_h, 512 * 512);
    cvt(W1, W1_h, 2048 * 512);
    cvt(W2, W2_h, 512 * 2048);

    // --- 2) gather / build fp16 inputs ---
    k_build_inputs<<<MT_, 256, 0, stream>>>(x, edgeattr, ts, W_temb, b_temb, eidx,
                                            nodecat, timecat, edgein);

    dim3 blk(128);
    dim3 g512(C_ / 256, MT_ / 16);        // (2, 4608)
    dim3 g2048(2048 / 256, MT_ / 16);     // (8, 4608)

    // --- 3) input projections ---
    k_gemm_wmma<EPI_F32 | EPI_F16><<<g512, blk, 0, stream>>>(
        nodecat, Wnode_h, b_node, C_, 512, nodef_f, nodef_h, nullptr);
    k_gemm_wmma<EPI_F16><<<g512, blk, 0, stream>>>(
        timecat, Wtimef_h, b_timef, C_, 128, nullptr, timef_h, nullptr);
    k_gemm_wmma<EPI_F16><<<g512, blk, 0, stream>>>(
        edgein, Wedge_h, b_edge, C_, 128, nullptr, edgef_h, nullptr);

    // --- 4) q/k/v ---
    k_gemm_wmma<EPI_F16><<<g512, blk, 0, stream>>>(
        nodef_h, Wq_h, bq, C_, 512, nullptr, q_h, nullptr);
    k_gemm_wmma<EPI_F16><<<g512, blk, 0, stream>>>(
        timef_h, Wk_h, bk, C_, 512, nullptr, k_h, nullptr);
    k_gemm_wmma<EPI_F16><<<g512, blk, 0, stream>>>(
        edgef_h, Wv_h, bv, C_, 512, nullptr, v_h, nullptr);

    // --- 5) 4x4 batch attention per (edge, head) ---
    k_edge_attention<<<LED_, 256, 0, stream>>>(q_h, k_h, v_h, ctx_h);

    // --- 6) attn_out + residual -> pre-LN, then LN1 ---
    k_gemm_wmma<EPI_RES | EPI_F32><<<g512, blk, 0, stream>>>(
        ctx_h, Wo_h, bo, C_, 512, out1_f, nullptr, nodef_f);
    k_layernorm<<<MT_ / 8, 256, 0, stream>>>(out1_f, g1, be1, out1_f, out1_h);

    // --- 7) FFN: gelu(out1 @ W1^T + b1) @ W2^T + b2 + out1, then LN2 ---
    k_gemm_wmma<EPI_GELU | EPI_F16><<<g2048, blk, 0, stream>>>(
        out1_h, W1_h, b1, 2048, 512, nullptr, ffn1_h, nullptr);
    k_gemm_wmma<EPI_RES | EPI_F32><<<g512, blk, 0, stream>>>(
        ffn1_h, W2_h, b2, C_, 2048, out, nullptr, out1_f);
    k_layernorm<<<MT_ / 8, 256, 0, stream>>>(out, g2, be2, out, nullptr);
}
